// MemorySelfAttention_77859167142132
// MI455X (gfx1250) — matrix operand order repaired and checked by reading, not compiled, not run on hardware
//
#include <hip/hip_runtime.h>
#include <math.h>

// ---------------- config (matches reference) ----------------
#define BB 2
#define TT 1024
#define CC 1024
#define HH 16
#define DD 64
#define SS 512
#define MM 1024
#define SEQ (SS + TT)          // 1536
#define TOT (MM + SS + TT)     // 2560
#define ROWS (BB * SEQ)        // 3072
#define N3 (3 * CC)            // 3072
#define SCALE 0.125f           // 1/sqrt(64)
#define LOG_THETA 9.210340371976184f  // ln(10000)

typedef __bf16 bf16;
typedef __bf16 v8bf  __attribute__((ext_vector_type(8)));
typedef __bf16 v16bf __attribute__((ext_vector_type(16)));
typedef float  v8f   __attribute__((ext_vector_type(8)));

#define WMMA_BF16(a, b, c) \
  __builtin_amdgcn_wmma_f32_16x16x32_bf16(false, (a), false, (b), (short)0, (c), false, false)

// load two 8xbf16 chunks (16B each) and fuse into one 16xbf16 fragment
__device__ __forceinline__ v16bf ld_pair(const bf16* p0, const bf16* p1) {
  v8bf lo = *reinterpret_cast<const v8bf*>(p0);
  v8bf hi = *reinterpret_cast<const v8bf*>(p1);
  return __builtin_shufflevector(lo, hi, 0,1,2,3,4,5,6,7,8,9,10,11,12,13,14,15);
}

__device__ __forceinline__ float half16_max(float x) {
  x = fmaxf(x, __shfl_xor(x, 1, 32));
  x = fmaxf(x, __shfl_xor(x, 2, 32));
  x = fmaxf(x, __shfl_xor(x, 4, 32));
  x = fmaxf(x, __shfl_xor(x, 8, 32));
  return x;
}

// async DMA of one 16B chunk per lane into LDS (ASYNCcnt-tracked)
__device__ __forceinline__ void async_ld16(unsigned ldsoff, unsigned long long gaddr) {
  asm volatile("global_load_async_to_lds_b128 %0, %1, off"
               :: "v"(ldsoff), "v"(gaddr) : "memory");
}

// ---------------- kernel 0: concat(short_mem, x) -> bf16 ----------------
__global__ __launch_bounds__(256) void msa_concat_bf16(const float* __restrict__ x,
                                                       const float* __restrict__ smem,
                                                       bf16* __restrict__ dst) {
  int idx = blockIdx.x * 256 + threadIdx.x;            // over ROWS*CC
  int col = idx & (CC - 1);
  int row = idx >> 10;
  int b = row / SEQ, p = row % SEQ;
  float v = (p < SS) ? smem[((size_t)b * SS + p) * CC + col]
                     : x[((size_t)b * TT + (p - SS)) * CC + col];
  dst[idx] = (bf16)v;
}

// ---------------- kernel 1: transpose fp32 (K x N) -> bf16 (N x K) ----------------
__global__ __launch_bounds__(256) void msa_transpose_bf16(const float* __restrict__ src,
                                                          bf16* __restrict__ dst,
                                                          int K, int N) {
  int idx = blockIdx.x * 256 + threadIdx.x;            // over N*K
  int k = idx % K;
  int n = idx / K;
  dst[idx] = (bf16)src[(size_t)k * N + n];
}

// ---------------- GEMM fragment helpers (ISA 7.12.2 layouts) ----------------
__device__ __forceinline__ void load_a4(v16bf av[4], const bf16* const arow[4], int koff, int lh) {
#pragma unroll
  for (int i = 0; i < 4; ++i)
    av[i] = ld_pair(arow[i] + koff + lh * 8, arow[i] + koff + 16 + lh * 8);
}
__device__ __forceinline__ void load_b4(v16bf bv[4], const bf16* const brow[4], int koff, int lh) {
#pragma unroll
  for (int j = 0; j < 4; ++j)
    bv[j] = ld_pair(brow[j] + koff + lh * 8, brow[j] + koff + 16 + lh * 8);
}
__device__ __forceinline__ void mma4x4(v8f acc[4][4], const v16bf av[4], const v16bf bv[4]) {
#pragma unroll
  for (int i = 0; i < 4; ++i)
#pragma unroll
    for (int j = 0; j < 4; ++j)
      acc[i][j] = WMMA_BF16(av[i], bv[j], acc[i][j]);
}

// ---------------- kernel 2/5: bf16 WMMA GEMM, C = A(MxK) * Bt(NxK)^T ----------------
// one wave computes a 64(M) x 64(N) tile: 16 WMMAs per K=32 step, double-buffered fragments
__global__ __launch_bounds__(256) void msa_gemm_bf16(const bf16* __restrict__ A,
                                                     const bf16* __restrict__ Bt,
                                                     float* __restrict__ C,
                                                     int Mdim, int Ndim, int Kdim) {
  int lane = threadIdx.x & 31;
  int wid = (blockIdx.x * 256 + threadIdx.x) >> 5;
  int ntiles = Ndim >> 6;                 // 64-wide tiles in N
  int mt = wid / ntiles, nt = wid % ntiles;
  if (mt * 64 >= Mdim) return;
  int lr = lane & 15, lh = lane >> 4;

  const bf16* arow[4];
  const bf16* brow[4];
#pragma unroll
  for (int i = 0; i < 4; ++i) arow[i] = A + (size_t)(mt * 64 + i * 16 + lr) * Kdim;
#pragma unroll
  for (int j = 0; j < 4; ++j) brow[j] = Bt + (size_t)(nt * 64 + j * 16 + lr) * Kdim;

  v8f zero = {0.f, 0.f, 0.f, 0.f, 0.f, 0.f, 0.f, 0.f};
  v8f acc[4][4];
#pragma unroll
  for (int i = 0; i < 4; ++i)
#pragma unroll
    for (int j = 0; j < 4; ++j) acc[i][j] = zero;

  v16bf a0[4], b0[4], a1[4], b1[4];
  load_a4(a0, arow, 0, lh);
  load_b4(b0, brow, 0, lh);

  for (int k0 = 0; k0 < Kdim; k0 += 64) {
#pragma unroll
    for (int i = 0; i < 4; ++i) __builtin_prefetch(arow[i] + k0 + 256, 0, 3);
    load_a4(a1, arow, k0 + 32, lh);
    load_b4(b1, brow, k0 + 32, lh);
    mma4x4(acc, a0, b0);
    int kn = (k0 + 64 < Kdim) ? k0 + 64 : 0;  // clamp; dead loads DCE'd on last iter
    load_a4(a0, arow, kn, lh);
    load_b4(b0, brow, kn, lh);
    mma4x4(acc, a1, b1);
  }

#pragma unroll
  for (int i = 0; i < 4; ++i)
#pragma unroll
    for (int j = 0; j < 4; ++j)
#pragma unroll
      for (int vv = 0; vv < 8; ++vv)
        C[(size_t)(mt * 64 + i * 16 + vv + lh * 8) * Ndim + nt * 64 + j * 16 + lr] =
            acc[i][j][vv];
}

// ---------------- kernel 3: RoPE + scatter into Q / K / V^T (bf16) ----------------
__global__ __launch_bounds__(256) void msa_rope_scatter(const float* __restrict__ qkv,
                                                        const float* __restrict__ long_k,
                                                        const float* __restrict__ long_v,
                                                        bf16* __restrict__ Qs,   // (B,H,1024,64)
                                                        bf16* __restrict__ Kall, // (B,H,2560,64)
                                                        bf16* __restrict__ Vt) { // (B,H,64,2560)
  int idx = blockIdx.x * 256 + threadIdx.x;
  int i = idx & 31;
  int h = (idx >> 5) & (HH - 1);
  int p = (idx >> 9) % TOT;
  int b = idx / (32 * HH * TOT);
  int bh = b * HH + h;

  if (p < MM) {  // long-memory prefix (no RoPE)
    const float* lk = long_k + (((size_t)b * MM + p) * HH + h) * DD;
    const float* lv = long_v + (((size_t)b * MM + p) * HH + h) * DD;
    Kall[((size_t)bh * TOT + p) * DD + 2 * i]     = (bf16)lk[2 * i];
    Kall[((size_t)bh * TOT + p) * DD + 2 * i + 1] = (bf16)lk[2 * i + 1];
    Vt[((size_t)bh * DD + 2 * i) * TOT + p]       = (bf16)lv[2 * i];
    Vt[((size_t)bh * DD + 2 * i + 1) * TOT + p]   = (bf16)lv[2 * i + 1];
  } else {
    int pp = p - MM;  // position within concat(short,x); also the RoPE position
    const float* qrow = qkv + ((size_t)b * SEQ + pp) * N3 + h * DD;
    const float* krow = qrow + CC;
    const float* vrow = qrow + 2 * CC;
    float inv_freq = __expf(-((float)(2 * i) / (float)DD) * LOG_THETA);
    float ang = (float)pp * inv_freq;
    float sn, cs;
    __sincosf(ang, &sn, &cs);
    float k0 = krow[2 * i], k1 = krow[2 * i + 1];
    Kall[((size_t)bh * TOT + p) * DD + 2 * i]     = (bf16)(k0 * cs - k1 * sn);
    Kall[((size_t)bh * TOT + p) * DD + 2 * i + 1] = (bf16)(k0 * sn + k1 * cs);
    Vt[((size_t)bh * DD + 2 * i) * TOT + p]       = (bf16)vrow[2 * i];
    Vt[((size_t)bh * DD + 2 * i + 1) * TOT + p]   = (bf16)vrow[2 * i + 1];
    if (pp >= SS) {  // only queries that reach the output; fold softmax scale in
      int qr = pp - SS;
      float q0 = qrow[2 * i], q1 = qrow[2 * i + 1];
      Qs[((size_t)bh * TT + qr) * DD + 2 * i]     = (bf16)((q0 * cs - q1 * sn) * SCALE);
      Qs[((size_t)bh * TT + qr) * DD + 2 * i + 1] = (bf16)((q0 * sn + q1 * cs) * SCALE);
    }
  }
}

// ---------------- kernel 4: flash attention ----------------
// one wave = 16-query tile; K/V tiles double-buffered in LDS via async DMA
// (global_load_async_to_lds_b128 + s_wait_asynccnt); row sums via P x ones WMMA.
__global__ __launch_bounds__(64) void msa_attn(const bf16* __restrict__ Qs,
                                               const bf16* __restrict__ Kall,
                                               const bf16* __restrict__ Vt,
                                               bf16* __restrict__ Ybf) {  // (B*1024, 1024)
  __shared__ __align__(16) bf16 sK[2][2][32 * 64];  // [wave][buf]: 32 keys x 64 d
  __shared__ __align__(16) bf16 sV[2][2][64 * 32];  // [wave][buf]: 64 d-rows x 32 keys
  __shared__ __align__(16) bf16 sP[2][16 * 32];     // per-wave P staging
  int lane = threadIdx.x & 31;
  int widx = threadIdx.x >> 5;
  int wg = blockIdx.x * 2 + widx;        // 0..2047
  int b = wg >> 10;
  int h = (wg >> 6) & (HH - 1);
  int qt = wg & 63;
  int bh = b * HH + h;
  int lr = lane & 15, lh = lane >> 4;
  bf16* pl = sP[widx];

  const bf16* kglob = Kall + (size_t)bh * TOT * DD;  // rows: key, 64 elems
  const bf16* vglob = Vt + (size_t)bh * DD * TOT;    // rows: d, 2560 elems

  const bf16* qbase = Qs + ((size_t)bh * TT + qt * 16 + lr) * DD;
  v16bf qf0 = ld_pair(qbase + lh * 16, qbase + lh * 16 + 8);          // d 0..31
  v16bf qf1 = ld_pair(qbase + 32 + lh * 16, qbase + 40 + lh * 16);     // d 32..63

  v16bf ones;
#pragma unroll
  for (int e = 0; e < 16; ++e) ones[e] = (bf16)1.0f;

  const float NEG = -__builtin_inff();
  float mrow[8], lrow[8], alv[8];
  v8f zero = {0.f, 0.f, 0.f, 0.f, 0.f, 0.f, 0.f, 0.f};
  v8f acc[4] = {zero, zero, zero, zero};
#pragma unroll
  for (int vv = 0; vv < 8; ++vv) { mrow[vv] = NEG; lrow[vv] = 0.f; }

  // async-DMA one 32-key block (K: 4KB contiguous; V^T: 64 rows x 64B)
  auto issue_kv = [&](int buf, int kb) {
    unsigned kl = (unsigned)(uintptr_t)(&sK[widx][buf][0]) + lane * 16;
    unsigned long long kg =
        (unsigned long long)(uintptr_t)(kglob + (size_t)kb * DD) + (unsigned)(lane * 16);
    unsigned vl = (unsigned)(uintptr_t)(&sV[widx][buf][0]) + (lane >> 2) * 64 + (lane & 3) * 16;
    unsigned long long vg = (unsigned long long)(uintptr_t)(vglob + kb) +
                            (unsigned long long)(lane >> 2) * (TOT * 2) + (unsigned)((lane & 3) * 16);
#pragma unroll
    for (int t = 0; t < 8; ++t) {
      async_ld16(kl + t * 512, kg + (unsigned long long)t * 512);
      async_ld16(vl + t * 512, vg + (unsigned long long)t * 8 * (TOT * 2));
    }
  };

  int qpos_base = MM + SS + qt * 16;
  int kend = qpos_base + 16;  // max keys any row of this tile may see

  issue_kv(0, 0);
  int buf = 0;
  for (int kb = 0; kb < kend; kb += 32) {
    // issue next block's DMA (clamped redundant copy on last iter keeps the wait uniform)
    int kbn = (kb + 32 < kend) ? kb + 32 : 0;
    issue_kv(buf ^ 1, kbn);
    asm volatile("s_wait_asynccnt 16" ::: "memory");  // current block resident

    // ---- S = Q K^T for 32 keys (two 16x16 tiles), K from LDS ----
    const bf16* kr0 = &sK[widx][buf][lr * 64 + lh * 16];
    const bf16* kr1 = kr0 + 16 * 64;
    v8f s0 = zero, s1 = zero;
    s0 = WMMA_BF16(qf0, ld_pair(kr0, kr0 + 8), s0);
    s0 = WMMA_BF16(qf1, ld_pair(kr0 + 32, kr0 + 40), s0);
    s1 = WMMA_BF16(qf0, ld_pair(kr1, kr1 + 8), s1);
    s1 = WMMA_BF16(qf1, ld_pair(kr1 + 32, kr1 + 40), s1);

    // ---- masked online softmax update ----
#pragma unroll
    for (int vv = 0; vv < 8; ++vv) {
      int qpos = qpos_base + vv + lh * 8;
      float a0 = s0[vv]; if (kb + lr > qpos) a0 = NEG;
      float a1 = s1[vv]; if (kb + 16 + lr > qpos) a1 = NEG;
      float rmax = half16_max(fmaxf(a0, a1));
      float nm = fmaxf(mrow[vv], rmax);
      float al = __expf(mrow[vv] - nm);
      alv[vv] = al;
      float p0 = __expf(a0 - nm);
      float p1 = __expf(a1 - nm);
      mrow[vv] = nm;
      acc[0][vv] *= al; acc[1][vv] *= al; acc[2][vv] *= al; acc[3][vv] *= al;
      // stage P (C-layout -> LDS row-major 16x32) for A-fragment reload
      pl[(vv + lh * 8) * 32 + lr]      = (bf16)p0;
      pl[(vv + lh * 8) * 32 + 16 + lr] = (bf16)p1;
    }
    // A fragment of P (within-wave DS ordering guarantees store->load)
    v16bf pa = ld_pair(pl + lr * 32 + lh * 16, pl + lr * 32 + lh * 16 + 8);

    // ---- row sums via WMMA: rsum = P x ones (every column = row sum) ----
    v8f rsum = WMMA_BF16(pa, ones, zero);
#pragma unroll
    for (int vv = 0; vv < 8; ++vv) lrow[vv] = lrow[vv] * alv[vv] + rsum[vv];

    // ---- acc += P * V  (V^T rows from LDS) ----
#pragma unroll
    for (int j = 0; j < 4; ++j) {
      const bf16* vr = &sV[widx][buf][(j * 16 + lr) * 32 + lh * 16];
      acc[j] = WMMA_BF16(pa, ld_pair(vr, vr + 8), acc[j]);
    }
    buf ^= 1;
  }
  asm volatile("s_wait_asynccnt 0" ::: "memory");

  // ---- normalize + store y (B,Tq,H,D) flattened as (B*1024, 1024) bf16 ----
#pragma unroll
  for (int vv = 0; vv < 8; ++vv) {
    float inv = 1.0f / lrow[vv];
    size_t row = (size_t)b * TT + qt * 16 + vv + lh * 8;
    bf16* yb = Ybf + row * CC + h * DD + lr;
    yb[0]  = (bf16)(acc[0][vv] * inv);
    yb[16] = (bf16)(acc[1][vv] * inv);
    yb[32] = (bf16)(acc[2][vv] * inv);
    yb[48] = (bf16)(acc[3][vv] * inv);
  }
}

// ---------------- host launcher ----------------
extern "C" void kernel_launch(void* const* d_in, const int* in_sizes, int n_in,
                              void* d_out, int out_size, void* d_ws, size_t ws_size,
                              hipStream_t stream) {
  const float* x     = (const float*)d_in[0];
  const float* smem  = (const float*)d_in[1];
  // d_in[2] = long_q: provably unused for the returned slice (queries >= M+S only)
  const float* lk    = (const float*)d_in[3];
  const float* lv    = (const float*)d_in[4];
  const float* wqkv  = (const float*)d_in[5];
  const float* wproj = (const float*)d_in[6];
  float* out = (float*)d_out;

  char* ws = (char*)d_ws;
  bf16*  inp_bf  = (bf16*)(ws + 0);              // 3072x1024 bf16   (6.0 MB)
  bf16*  wqkv_t  = (bf16*)(ws + 6291456);        // 3072x1024 bf16   (6.0 MB)
  bf16*  wproj_t = (bf16*)(ws + 12582912);       // 1024x1024 bf16   (2.0 MB)
  float* qkv     = (float*)(ws + 14680064);      // 3072x3072 f32    (36 MB)
  bf16*  Qs      = (bf16*)(ws + 52428800);       // (B,H,1024,64)    (4.0 MB)
  bf16*  Kall    = (bf16*)(ws + 56623104);       // (B,H,2560,64)    (10 MB)
  bf16*  Vt      = (bf16*)(ws + 67108864);       // (B,H,64,2560)    (10 MB)
  bf16*  Ybf     = (bf16*)(ws + 77594624);       // 2048x1024 bf16   (4.0 MB)

  // 0: concat + convert input
  msa_concat_bf16<<<(ROWS * CC) / 256, 256, 0, stream>>>(x, smem, inp_bf);
  // 1: weight transposes
  msa_transpose_bf16<<<(N3 * CC) / 256, 256, 0, stream>>>(wqkv, wqkv_t, CC, N3);
  msa_transpose_bf16<<<(CC * CC) / 256, 256, 0, stream>>>(wproj, wproj_t, CC, CC);
  // 2: qkv = inp @ Wqkv   (M=3072, N=3072, K=1024): 48x48 = 2304 waves -> 288 blocks
  msa_gemm_bf16<<<((ROWS / 64) * (N3 / 64) * 32) / 256, 256, 0, stream>>>(
      inp_bf, wqkv_t, qkv, ROWS, N3, CC);
  // 3: RoPE + scatter (threads = B*TOT*H*32)
  msa_rope_scatter<<<(BB * TOT * HH * 32) / 256, 256, 0, stream>>>(qkv, lk, lv, Qs, Kall, Vt);
  // 4: flash attention (2048 waves, 2 waves/block, async-LDS double buffering)
  msa_attn<<<(BB * HH * (TT / 16)) / 2, 64, 0, stream>>>(Qs, Kall, Vt, Ybf);
  // 5: out = y @ Wproj    (M=2048, N=1024, K=1024): 32x16 = 512 waves -> 64 blocks
  msa_gemm_bf16<<<((BB * TT / 64) * (CC / 64) * 32) / 256, 256, 0, stream>>>(
      Ybf, wproj_t, out, BB * TT, CC, CC);
}
